// Encoder_3521873183605
// MI455X (gfx1250) — matrix operand order repaired and checked by reading, not compiled, not run on hardware
//
#include <hip/hip_runtime.h>
#include <math.h>

#define NSEQ 3072
#define EMB  768
#define NH   12
#define HD   64
#define NLAY 4
#define FFD  3072

typedef __attribute__((ext_vector_type(16))) __bf16        v16bf;
typedef __attribute__((ext_vector_type(8)))  float         v8f;
typedef __attribute__((ext_vector_type(4)))  unsigned int  u32x4;
typedef __attribute__((ext_vector_type(8)))  int           i32x8;
typedef __attribute__((ext_vector_type(4)))  int           i32x4;

__device__ __forceinline__ unsigned short f2bf(float f) {
  unsigned int u = __float_as_uint(f);
  u += 0x7FFFu + ((u >> 16) & 1u);          // round-to-nearest-even
  return (unsigned short)(u >> 16);
}

union ABFrag {
  v16bf v;
  u32x4 q[2];
  unsigned int d[8];
  unsigned short u[16];
};

// low 32 bits of a flat shared pointer == workgroup-relative LDS byte offset
__device__ __forceinline__ unsigned lds_off(const void* p) {
  return (unsigned)reinterpret_cast<unsigned long long>(p);
}

// ---------------------------------------------------------------------------
// TDM: async 2-D tile (rows x rowlen bf16, row stride in elements) -> LDS.
// D# per CDNA5 ISA ch.8: group0 {count=1, lds_addr, global_addr, type=2},
// group1 {data_size=2B, tensor_dim0/1, tile_dim0/1, tensor_dim0_stride}.
// Groups 2/3 zero (2-D). Tracked with TENSORcnt.
// ---------------------------------------------------------------------------
__device__ __forceinline__ void tdm_load_2d(unsigned lds_addr,
                                            const unsigned short* gptr,
                                            unsigned rows, unsigned rowlen,
                                            unsigned long long row_stride_elems)
{
  const unsigned long long ga = reinterpret_cast<unsigned long long>(gptr);
  u32x4 g0;
  g0.x = 1u;                                              // count=1, user mode
  g0.y = lds_addr;                                        // [63:32] lds_addr
  g0.z = (unsigned)ga;                                    // global_addr[31:0]
  g0.w = (unsigned)((ga >> 32) & 0x01FFFFFFu) | (2u << 30); // addr[56:32] | type=2
  i32x8 g1;
  g1[0] = (int)(1u << 16);                                // data_size=1 (2 bytes)
  g1[1] = (int)((rowlen & 0xFFFFu) << 16);                // tensor_dim0[15:0]
  g1[2] = (int)((rows & 0xFFFFu) << 16);                  // td0 hi=0 | tensor_dim1[15:0]
  g1[3] = (int)((rowlen & 0xFFFFu) << 16);                // td1 hi=0 | tile_dim0
  g1[4] = (int)(rows & 0xFFFFu);                          // tile_dim1 | tile_dim2=0
  g1[5] = (int)(row_stride_elems & 0xFFFFFFFFu);          // tensor_dim0_stride lo
  g1[6] = (int)((row_stride_elems >> 32) & 0xFFFFu);      // stride hi | dim1_stride lo=0
  g1[7] = 0;
  i32x4 z4 = {0, 0, 0, 0};
#if defined(__clang_major__) && (__clang_major__ >= 23)
  i32x8 z8 = {0, 0, 0, 0, 0, 0, 0, 0};
  __builtin_amdgcn_tensor_load_to_lds(g0, g1, z4, z4, z8, 0);
#else
  __builtin_amdgcn_tensor_load_to_lds(g0, g1, z4, z4, 0);
#endif
}

// ---------------------------------------------------------------------------
// C = act( A @ B + bias ), bf16 operands, fp32 WMMA accumulation.
//   A : M x K row-major (lda), bf16 as ushort
//   B : element (k,n) at B[n*ldb + k]  (rows contiguous in k)
//   C : f32 or bf16 (OUTBF); TRANSC stores C^T (C[col*ldc + row])
// Block: 8 waves = WM x WN wave grid, each wave a 32x32 tile (2x2 WMMA).
// Block tile (WM*32) x (WN*32). A/B tiles staged in LDS by the Tensor Data
// Mover, double-buffered, TENSORcnt-synchronized. K multiple of 32.
// ---------------------------------------------------------------------------
template<int BIAS, int RELU, int OUTBF, int TRANSC, int WM, int WN>
__global__ void __launch_bounds__(256)
gemm_tdm_wmma(const unsigned short* __restrict__ A, int lda,
              const unsigned short* __restrict__ B, int ldb,
              const float* __restrict__ bias,
              void* __restrict__ Cp, long ldc,
              int M, int Ncols, int K)
{
  (void)M; (void)Ncols;
  __shared__ __align__(16) unsigned short As[2][WM * 32 * 32];
  __shared__ __align__(16) unsigned short Bs[2][WN * 32 * 32];

  const int lane = threadIdx.x;                 // 0..31 (wave32)
  const int w    = threadIdx.y;                 // 0..7
  const int r  = lane & 15;
  const int hi = lane >> 4;
  const int m0 = blockIdx.y * (WM * 32);
  const int n0 = blockIdx.x * (WN * 32);
  const int wm = (w % WM) * 32;
  const int wn = (w / WM) * 32;

  v8f acc[2][2] = {};

  const unsigned short* Ag = A + (long)m0 * lda;
  const unsigned short* Bg = B + (long)n0 * ldb;
  const int nsteps = K >> 5;

  if (w == 0) {
    tdm_load_2d(lds_off(As[0]), Ag, WM * 32, 32, (unsigned long long)lda);
    tdm_load_2d(lds_off(Bs[0]), Bg, WN * 32, 32, (unsigned long long)ldb);
  }

  for (int s = 0; s < nsteps; ++s) {
    const int cur = s & 1;
    if (w == 0) __builtin_amdgcn_s_wait_tensorcnt(0);
    __syncthreads();                            // tile `cur` visible to all waves
    if (w == 0 && (s + 1) < nsteps) {           // prefetch next tile into other buffer
      const long k0n = (long)(s + 1) << 5;
      tdm_load_2d(lds_off(As[cur ^ 1]), Ag + k0n, WM * 32, 32, (unsigned long long)lda);
      tdm_load_2d(lds_off(Bs[cur ^ 1]), Bg + k0n, WN * 32, 32, (unsigned long long)ldb);
    }

    const unsigned short* as = As[cur];
    const unsigned short* bs = Bs[cur];
    ABFrag a[2], b[2];
#pragma unroll
    for (int i = 0; i < 2; ++i) {
      // A 16x32 bf16 frag: lane holds K {hi*8..+7} and {16+hi*8..+7} of its row
      const unsigned short* rp = as + (wm + i * 16 + r) * 32;
      a[i].q[0] = *(const u32x4*)(rp + hi * 8);
      a[i].q[1] = *(const u32x4*)(rp + 16 + hi * 8);
    }
#pragma unroll
    for (int j = 0; j < 2; ++j) {
      // B 32x16 bf16 frag: lane holds K {hi*8..+7} and {16+hi*8..+7} of its column
      const unsigned short* cp = bs + (wn + j * 16 + r) * 32;
      b[j].q[0] = *(const u32x4*)(cp + hi * 8);
      b[j].q[1] = *(const u32x4*)(cp + 16 + hi * 8);
    }
#pragma unroll
    for (int i = 0; i < 2; ++i)
#pragma unroll
      for (int j = 0; j < 2; ++j)
        acc[i][j] = __builtin_amdgcn_wmma_f32_16x16x32_bf16(
            false, a[i].v, false, b[j].v, (short)0, acc[i][j], false, false);
  }

#pragma unroll
  for (int j = 0; j < 2; ++j) {
    const int col = n0 + wn + j * 16 + r;
    float bv = 0.0f;
    if (BIAS) bv = bias[col];
#pragma unroll
    for (int i = 0; i < 2; ++i) {
#pragma unroll
      for (int ii = 0; ii < 8; ++ii) {
        const long row = m0 + wm + i * 16 + ii + 8 * hi;  // C layout: VGPR ii -> M = ii+8*hi
        float v = acc[i][j][ii] + bv;
        if (RELU) v = fmaxf(v, 0.0f);
        if (TRANSC)     ((unsigned short*)Cp)[(long)col * ldc + row] = f2bf(v);
        else if (OUTBF) ((unsigned short*)Cp)[row * ldc + col] = f2bf(v);
        else            ((float*)Cp)[row * ldc + col] = v;
      }
    }
  }
}

// ---------------------------------------------------------------------------
// Reductions (wave32)
// ---------------------------------------------------------------------------
__device__ __forceinline__ float warp_sum(float v) {
  for (int o = 16; o > 0; o >>= 1) v += __shfl_down(v, o, 32);
  return v;
}
__device__ __forceinline__ float warp_max(float v) {
  for (int o = 16; o > 0; o >>= 1) v = fmaxf(v, __shfl_down(v, o, 32));
  return v;
}
__device__ __forceinline__ float block_sum(float v, float* sh) {   // sh[>=9]
  const int lane = threadIdx.x & 31, wv = threadIdx.x >> 5;
  v = warp_sum(v);
  if (lane == 0) sh[wv] = v;
  __syncthreads();
  if (threadIdx.x < 32) {
    float t = (threadIdx.x < (blockDim.x >> 5)) ? sh[threadIdx.x] : 0.0f;
    t = warp_sum(t);
    if (threadIdx.x == 0) sh[8] = t;
  }
  __syncthreads();
  float rr = sh[8];
  __syncthreads();
  return rr;
}
__device__ __forceinline__ float block_max(float v, float* sh) {
  const int lane = threadIdx.x & 31, wv = threadIdx.x >> 5;
  v = warp_max(v);
  if (lane == 0) sh[wv] = v;
  __syncthreads();
  if (threadIdx.x < 32) {
    float t = (threadIdx.x < (blockDim.x >> 5)) ? sh[threadIdx.x] : -3.4e38f;
    t = warp_max(t);
    if (threadIdx.x == 0) sh[8] = t;
  }
  __syncthreads();
  float rr = sh[8];
  __syncthreads();
  return rr;
}

// ---------------------------------------------------------------------------
// Row softmax of (scores * inv_scale), bf16 output. One block per row.
// ---------------------------------------------------------------------------
__global__ void softmax_rows(const float* __restrict__ S, unsigned short* __restrict__ P,
                             int cols, float inv_scale)
{
  __shared__ float buf[NSEQ];
  __shared__ float red[9];
  const long row = blockIdx.x;
  const float* s = S + row * (long)cols;
  unsigned short* p = P + row * (long)cols;

  float mx = -3.4e38f;
  for (int i = threadIdx.x; i < cols; i += blockDim.x) {
    float v = s[i] * inv_scale;
    buf[i] = v;
    mx = fmaxf(mx, v);
  }
  mx = block_max(mx, red);

  float sum = 0.0f;
  for (int i = threadIdx.x; i < cols; i += blockDim.x) {
    float e = __expf(buf[i] - mx);
    buf[i] = e;
    sum += e;
  }
  sum = block_sum(sum, red);
  const float rs = 1.0f / sum;
  for (int i = threadIdx.x; i < cols; i += blockDim.x)
    p[i] = f2bf(buf[i] * rs);
}

// ---------------------------------------------------------------------------
// out = LayerNorm(A + B) * g + beta ; writes f32 and bf16. One block per row.
// ---------------------------------------------------------------------------
__global__ void add_layernorm(const float* __restrict__ A, const float* __restrict__ B,
                              const float* __restrict__ g, const float* __restrict__ beta,
                              float* __restrict__ outf, unsigned short* __restrict__ outb,
                              int cols)
{
  __shared__ float buf[EMB];
  __shared__ float red[9];
  const long row = blockIdx.x;
  const long base = row * (long)cols;

  float s = 0.0f;
  for (int i = threadIdx.x; i < cols; i += blockDim.x) {
    float v = A[base + i] + B[base + i];
    buf[i] = v;
    s += v;
  }
  s = block_sum(s, red);
  const float mu = s / (float)cols;

  float ss = 0.0f;
  for (int i = threadIdx.x; i < cols; i += blockDim.x) {
    float d = buf[i] - mu;
    ss += d * d;
  }
  ss = block_sum(ss, red);
  const float rstd = rsqrtf(ss / (float)cols + 1e-5f);

  for (int i = threadIdx.x; i < cols; i += blockDim.x) {
    float v = (buf[i] - mu) * rstd * g[i] + beta[i];
    outf[base + i] = v;
    outb[base + i] = f2bf(v);
  }
}

// ---------------------------------------------------------------------------
// Elementwise helpers
// ---------------------------------------------------------------------------
__global__ void f32_to_bf16_k(const float* __restrict__ src, unsigned short* __restrict__ dst, long n) {
  long i = (long)blockIdx.x * blockDim.x + threadIdx.x;
  const long stride = (long)gridDim.x * blockDim.x;
  for (; i < n; i += stride) dst[i] = f2bf(src[i]);
}

__global__ void trunc_add_pos(const float* __restrict__ t, const float* __restrict__ pos,
                              float* __restrict__ outf, unsigned short* __restrict__ outb, long n) {
  long i = (long)blockIdx.x * blockDim.x + threadIdx.x;
  const long stride = (long)gridDim.x * blockDim.x;
  for (; i < n; i += stride) {
    float v = truncf(t[i]) + pos[i];
    outf[i] = v;
    outb[i] = f2bf(v);
  }
}

__global__ void mean_cols(const float* __restrict__ X, float* __restrict__ out,
                          int rows, int cols) {
  const int c = blockIdx.x * blockDim.x + threadIdx.x;
  if (c >= cols) return;
  float s = 0.0f;
  for (int rr = 0; rr < rows; ++rr) s += X[(long)rr * cols + c];
  out[c] = s / (float)rows;
}

// ---------------------------------------------------------------------------
// Launcher
// ---------------------------------------------------------------------------
extern "C" void kernel_launch(void* const* d_in, const int* in_sizes, int n_in,
                              void* d_out, int out_size, void* d_ws, size_t ws_size,
                              hipStream_t stream) {
  (void)in_sizes; (void)n_in; (void)out_size; (void)ws_size;

  const float* x      = (const float*)d_in[0];
  // d_in[1] = mask : has no effect (reference discards the masked_fill result)
  const float* W_word = (const float*)d_in[2];
  const float* b_word = (const float*)d_in[3];
  const float* pos    = (const float*)d_in[4];
  const float* Wq     = (const float*)d_in[5];
  const float* Wk     = (const float*)d_in[6];
  const float* Wv     = (const float*)d_in[7];
  const float* Wo     = (const float*)d_in[8];
  const float* W1     = (const float*)d_in[9];
  const float* b1     = (const float*)d_in[10];
  const float* W2     = (const float*)d_in[11];
  const float* b2     = (const float*)d_in[12];
  const float* g1     = (const float*)d_in[13];
  const float* be1    = (const float*)d_in[14];
  const float* g2     = (const float*)d_in[15];
  const float* be2    = (const float*)d_in[16];
  float* out_final    = (float*)d_out;

  const long NE  = (long)NSEQ * EMB;
  const long NN  = (long)NSEQ * NSEQ;
  const long NF  = (long)NSEQ * FFD;
  const long WEE = (long)NLAY * EMB * EMB;
  const long WFE = (long)NLAY * FFD * EMB;

  char* p = (char*)d_ws;
  auto carve = [&](size_t bytes) -> char* {
    char* rr = p;
    p += (bytes + 255) & ~(size_t)255;
    return rr;
  };
  unsigned short* wwbf = (unsigned short*)carve((size_t)EMB * 768 * 2);
  unsigned short* wqbf = (unsigned short*)carve((size_t)WEE * 2);
  unsigned short* wkbf = (unsigned short*)carve((size_t)WEE * 2);
  unsigned short* wvbf = (unsigned short*)carve((size_t)WEE * 2);
  unsigned short* wobf = (unsigned short*)carve((size_t)WEE * 2);
  unsigned short* w1bf = (unsigned short*)carve((size_t)WFE * 2);
  unsigned short* w2bf = (unsigned short*)carve((size_t)WFE * 2);
  float* outF   = (float*)carve((size_t)NE * 4);
  float* xmF    = (float*)carve((size_t)NE * 4);
  float* t0F    = (float*)carve((size_t)NE * 4);
  float* scores = (float*)carve((size_t)NN * 4);     // also reused as f32 N*E temp
  unsigned short* xbf  = (unsigned short*)carve((size_t)NE * 2);
  unsigned short* obf  = (unsigned short*)carve((size_t)NE * 2);
  unsigned short* xmbf = (unsigned short*)carve((size_t)NE * 2);
  unsigned short* qbf  = (unsigned short*)carve((size_t)NE * 2);
  unsigned short* kbf  = (unsigned short*)carve((size_t)NE * 2);
  unsigned short* vtbf = (unsigned short*)carve((size_t)NE * 2);  // V^T, [E][NSEQ]
  unsigned short* pbf  = (unsigned short*)carve((size_t)NN * 2);
  unsigned short* ffbf = (unsigned short*)carve((size_t)NF * 2);

  const dim3 blk(32, 8);
  auto grid24 = [](int M, int Nc) { return dim3((unsigned)(Nc / 128), (unsigned)(M / 64)); };
  auto grid42 = [](int M, int Nc) { return dim3((unsigned)(Nc / 64), (unsigned)(M / 128)); };

  auto cvt = [&](const float* src, unsigned short* dst, long n) {
    int blocks = (int)((n + 1023) / 1024);
    if (blocks > 4096) blocks = 4096;
    f32_to_bf16_k<<<blocks, 256, 0, stream>>>(src, dst, n);
  };

  // bf16 conversions (weights once per call, activations as produced)
  cvt(x, xbf, NE);
  cvt(W_word, wwbf, (long)EMB * 768);
  cvt(Wq, wqbf, WEE);
  cvt(Wk, wkbf, WEE);
  cvt(Wv, wvbf, WEE);
  cvt(Wo, wobf, WEE);
  cvt(W1, w1bf, WFE);
  cvt(W2, w2bf, WFE);

  // word encoding: t0 = x @ W_word^T + b_word
  gemm_tdm_wmma<1, 0, 0, 0, 2, 4><<<grid24(NSEQ, EMB), blk, 0, stream>>>(
      xbf, 768, wwbf, 768, b_word, t0F, (long)EMB, NSEQ, EMB, 768);

  // out = trunc(t0) + pos_emb ; obf = bf16(out)
  trunc_add_pos<<<2304, 256, 0, stream>>>(t0F, pos, outF, obf, NE);

  const float inv_scale = 1.0f / sqrtf((float)EMB);   // softmax(energy / sqrt(E))

  for (int l = 0; l < NLAY; ++l) {
    const unsigned short* wq_l = wqbf + (long)l * EMB * EMB;
    const unsigned short* wk_l = wkbf + (long)l * EMB * EMB;
    const unsigned short* wv_l = wvbf + (long)l * EMB * EMB;
    const unsigned short* wo_l = wobf + (long)l * EMB * EMB;
    const unsigned short* w1_l = w1bf + (long)l * FFD * EMB;
    const unsigned short* w2_l = w2bf + (long)l * EMB * FFD;

    // Q, K row-major bf16 ; V stored transposed (per-head V^T rows contiguous)
    gemm_tdm_wmma<0, 0, 1, 0, 2, 4><<<grid24(NSEQ, EMB), blk, 0, stream>>>(
        obf, EMB, wq_l, EMB, nullptr, qbf, (long)EMB, NSEQ, EMB, EMB);
    gemm_tdm_wmma<0, 0, 1, 0, 2, 4><<<grid24(NSEQ, EMB), blk, 0, stream>>>(
        obf, EMB, wk_l, EMB, nullptr, kbf, (long)EMB, NSEQ, EMB, EMB);
    gemm_tdm_wmma<0, 0, 1, 1, 2, 4><<<grid24(NSEQ, EMB), blk, 0, stream>>>(
        obf, EMB, wv_l, EMB, nullptr, vtbf, (long)NSEQ, NSEQ, EMB, EMB);

    for (int h = 0; h < NH; ++h) {
      const long ho = (long)h * HD;
      // S = Q_h @ K_h^T  [N x N], K = 64
      gemm_tdm_wmma<0, 0, 0, 0, 2, 4><<<grid24(NSEQ, NSEQ), blk, 0, stream>>>(
          qbf + ho, EMB, kbf + ho, EMB, nullptr, scores, (long)NSEQ, NSEQ, NSEQ, HD);
      // P = softmax(S / sqrt(E)) -> bf16
      softmax_rows<<<NSEQ, 256, 0, stream>>>(scores, pbf, NSEQ, inv_scale);
      // t0[:, h*64:(h+1)*64] = P @ V_h  (B rows = V^T rows, k-contiguous)
      gemm_tdm_wmma<0, 0, 0, 0, 4, 2><<<grid42(NSEQ, HD), blk, 0, stream>>>(
          pbf, NSEQ, vtbf + ho * NSEQ, NSEQ, nullptr,
          (void*)(t0F + ho), (long)EMB, NSEQ, HD, NSEQ);
    }

    // O projection (reuse scores buffer as f32 N*E temp)
    cvt(t0F, xbf, NE);
    float* tproj = scores;
    gemm_tdm_wmma<0, 0, 0, 0, 2, 4><<<grid24(NSEQ, EMB), blk, 0, stream>>>(
        xbf, EMB, wo_l, EMB, nullptr, tproj, (long)EMB, NSEQ, EMB, EMB);

    // xm = LN(tproj + out)
    add_layernorm<<<NSEQ, 256, 0, stream>>>(tproj, outF, g1 + (long)l * EMB,
                                            be1 + (long)l * EMB, xmF, xmbf, EMB);

    // ff = relu(xm @ W1^T + b1) -> bf16
    gemm_tdm_wmma<1, 1, 1, 0, 2, 4><<<grid24(NSEQ, FFD), blk, 0, stream>>>(
        xmbf, EMB, w1_l, EMB, b1 + (long)l * FFD, ffbf, (long)FFD, NSEQ, FFD, EMB);

    // y = ff @ W2^T + b2 -> f32
    gemm_tdm_wmma<1, 0, 0, 0, 2, 4><<<grid24(NSEQ, EMB), blk, 0, stream>>>(
        ffbf, FFD, w2_l, FFD, b2 + (long)l * EMB, t0F, (long)EMB, NSEQ, EMB, FFD);

    // out = LN(xm + y)
    add_layernorm<<<NSEQ, 256, 0, stream>>>(xmF, t0F, g2 + (long)l * EMB,
                                            be2 + (long)l * EMB, outF, obf, EMB);
  }

  // final: mean over sequence -> [1,1,E]
  mean_cols<<<(EMB + 255) / 256, 256, 0, stream>>>(outF, out_final, NSEQ, EMB);
}
